// MultiLatentHeadAttention_54202487275615
// MI455X (gfx1250) — matrix operand order repaired and checked by reading, not compiled, not run on hardware
//
#include <hip/hip_runtime.h>

typedef unsigned short ushort_t;
typedef unsigned int   uint32;
typedef __attribute__((ext_vector_type(16))) __bf16 v16bf;
typedef __attribute__((ext_vector_type(8)))  float  v8f;

// ---------------------------------------------------------------- helpers
__device__ __forceinline__ ushort_t f32_to_bf16(float f) {
  union { float f; uint32 u; } v; v.f = f;
  uint32 r = v.u + 0x7FFFu + ((v.u >> 16) & 1u);   // round-to-nearest-even
  return (ushort_t)(r >> 16);
}

// ---------------------------------------------------------------- convert
__global__ __launch_bounds__(256) void k_cvt_bf16(const float* __restrict__ in,
                                                  ushort_t* __restrict__ out, long n) {
  long i = (long)blockIdx.x * blockDim.x + threadIdx.x;
  long stride = (long)gridDim.x * blockDim.x;
  for (; i < n; i += stride) out[i] = f32_to_bf16(in[i]);
}

// W[K,N] f32 -> Wt[N,K] bf16   (K,N multiples of 32)
__global__ __launch_bounds__(256) void k_transpose_cvt(const float* __restrict__ W,
                                                       ushort_t* __restrict__ Wt,
                                                       int K, int N) {
  __shared__ ushort_t t[32][33];
  int k0 = blockIdx.y * 32, n0 = blockIdx.x * 32;
  for (int i = threadIdx.y; i < 32; i += 8)
    t[i][threadIdx.x] = f32_to_bf16(W[(long)(k0 + i) * N + (n0 + threadIdx.x)]);
  __syncthreads();
  for (int i = threadIdx.y; i < 32; i += 8)
    Wt[(long)(n0 + i) * K + (k0 + threadIdx.x)] = t[threadIdx.x][i];
}

// ---------------------------------------------------------------- softmax
// one block per row; f32 scores -> bf16 probs
__global__ __launch_bounds__(128) void k_softmax_bf16(const float* __restrict__ S,
                                                      ushort_t* __restrict__ P, int C) {
  long base = (long)blockIdx.x * C;
  int tid = threadIdx.x;
  __shared__ float red[128];
  float mx = -3.4e38f;
  for (int c = tid; c < C; c += 128) mx = fmaxf(mx, S[base + c]);
  red[tid] = mx; __syncthreads();
  for (int s = 64; s > 0; s >>= 1) { if (tid < s) red[tid] = fmaxf(red[tid], red[tid + s]); __syncthreads(); }
  mx = red[0]; __syncthreads();
  float sum = 0.f;
  for (int c = tid; c < C; c += 128) sum += __expf(S[base + c] - mx);
  red[tid] = sum; __syncthreads();
  for (int s = 64; s > 0; s >>= 1) { if (tid < s) red[tid] += red[tid + s]; __syncthreads(); }
  float inv = 1.0f / red[0];
  for (int c = tid; c < C; c += 128) P[base + c] = f32_to_bf16(__expf(S[base + c] - mx) * inv);
}

// ---------------------------------------------------------------- GEMM
// C[m,n] = alpha * sum_k A[m*lda+k] * Bt[n*ldb+k]
// batched over grid.z: b = z/nH, h = z%nH ; per-operand (b,h) element strides.
// flags: bit0 = bf16 output, bit1 = blocked-transposed store:
//   addr = coff + (m/Mblock)*blockStride + n*ldct + (m%Mblock)
// Requirements: K % 32 == 0, N % 64 == 0, 16B-aligned A/Bt row starts.
union FragAB { v16bf v; uint32 u[8]; };
union FragC  { v8f  v; float  f[8]; };

#define LDS_STRIDE 40   // 80B rows: 16B-aligned for async B128 LDS writes

__global__ __launch_bounds__(256) void k_gemm_bf16(
    const ushort_t* __restrict__ A, int lda, long sAb, long sAh,
    const ushort_t* __restrict__ B, int ldb, long sBb, long sBh,
    void* __restrict__ C, int ldc, long sCb, long sCh,
    int M, int N, int K, float alpha, int nH, int flags,
    int Mblock, long blockStride, int ldct)
{
  __shared__ ushort_t lA[2][128][LDS_STRIDE];
  __shared__ ushort_t lB[2][64][LDS_STRIDE];

  const int tid  = threadIdx.x;
  const int lane = tid & 31;
  const int wave = tid >> 5;
  const int wm = wave & 3;     // M direction: 4 waves * 32 rows
  const int wn = wave >> 2;    // N direction: 2 waves * 32 cols

  const int bz = blockIdx.z;
  const int bb = bz / nH, hh = bz % nH;
  const ushort_t* Ab = A + (long)bb * sAb + (long)hh * sAh;
  const ushort_t* Bb = B + (long)bb * sBb + (long)hh * sBh;

  const int m0 = blockIdx.y * 128;
  const int n0 = blockIdx.x * 64;

  // ---- per-thread staging role (wave-uniform: waves 0-3 -> A, waves 4-5 -> B)
  const ushort_t* grow = nullptr;      // global row base
  unsigned l0 = 0, l1 = 0;             // LDS dst offsets (both buffers)
  if (tid < 128) {
    int gm = m0 + tid;
    if (gm >= M) gm = M - 1;           // clamp: garbage only feeds masked-out C rows
    grow = Ab + (long)gm * lda;
    l0 = (unsigned)(size_t)&lA[0][tid][0];
    l1 = (unsigned)(size_t)&lA[1][tid][0];
  } else if (tid < 192) {
    int r = tid - 128;
    grow = Bb + (long)(n0 + r) * ldb;  // always in range (N % 64 == 0)
    l0 = (unsigned)(size_t)&lB[0][r][0];
    l1 = (unsigned)(size_t)&lB[1][r][0];
  }

  // issue one 64B tile row as 4 async direct-to-LDS B128 loads
  // (INST_OFFSET is added to BOTH the global and the LDS address)
  auto issue = [&](int k0, int buf) {
    if (tid < 192) {
      unsigned l = buf ? l1 : l0;
      const ushort_t* g = grow + k0;
      asm volatile(
        "global_load_async_to_lds_b128 %0, %1, off\n\t"
        "global_load_async_to_lds_b128 %0, %1, off offset:16\n\t"
        "global_load_async_to_lds_b128 %0, %1, off offset:32\n\t"
        "global_load_async_to_lds_b128 %0, %1, off offset:48"
        :: "v"(l), "v"(g) : "memory");
    }
  };

  v8f acc[2][2];
  #pragma unroll
  for (int i = 0; i < 2; ++i)
    #pragma unroll
    for (int j = 0; j < 2; ++j)
      #pragma unroll
      for (int r = 0; r < 8; ++r) acc[i][j][r] = 0.0f;

  const int nk = K >> 5;
  issue(0, 0);

  for (int it = 0; it < nk; ++it) {
    const int cur = it & 1;
    if (it + 1 < nk) {
      issue((it + 1) << 5, cur ^ 1);
      // async loads complete in order: <=4 outstanding => current tile landed
      asm volatile("s_wait_asynccnt 0x4" ::: "memory");
    } else {
      asm volatile("s_wait_asynccnt 0x0" ::: "memory");
    }
    __syncthreads();

    // ---- fragments (ISA VGPR layouts, wave32)
    const int khalf = lane >> 4;
    const int l16   = lane & 15;
    FragAB fa[2], fb[2];
    #pragma unroll
    for (int i = 0; i < 2; ++i) {
      const ushort_t* rp = &lA[cur][wm*32 + i*16 + l16][8*khalf];
      #pragma unroll
      for (int p = 0; p < 4; ++p) fa[i].u[p]     = *(const uint32*)(rp + 2*p);
      #pragma unroll
      for (int p = 0; p < 4; ++p) fa[i].u[4 + p] = *(const uint32*)(rp + 16 + 2*p);
    }
    #pragma unroll
    for (int j = 0; j < 2; ++j) {
      const ushort_t* rp = &lB[cur][wn*32 + j*16 + l16][8*khalf];
      #pragma unroll
      for (int p = 0; p < 4; ++p) fb[j].u[p]     = *(const uint32*)(rp + 2*p);
      #pragma unroll
      for (int p = 0; p < 4; ++p) fb[j].u[4 + p] = *(const uint32*)(rp + 16 + 2*p);
    }

    #pragma unroll
    for (int i = 0; i < 2; ++i)
      #pragma unroll
      for (int j = 0; j < 2; ++j)
        acc[i][j] = __builtin_amdgcn_wmma_f32_16x16x32_bf16(
            false, fa[i].v, false, fb[j].v, (short)0, acc[i][j], false, false);

    __syncthreads();   // protect buffer about to be overwritten by next issue
  }

  // ---- epilogue
  const long coff = (long)bb * sCb + (long)hh * sCh;
  const int l16 = lane & 15, lhi = lane >> 4;
  #pragma unroll
  for (int i = 0; i < 2; ++i) {
    #pragma unroll
    for (int j = 0; j < 2; ++j) {
      FragC fc; fc.v = acc[i][j];
      int nn = n0 + wn*32 + j*16 + l16;
      #pragma unroll
      for (int r = 0; r < 8; ++r) {
        int mm = m0 + wm*32 + i*16 + lhi*8 + r;
        if (mm < M) {
          float val = fc.f[r] * alpha;
          long addr;
          if (flags & 2) addr = coff + (long)(mm / Mblock) * blockStride + (long)nn * ldct + (mm % Mblock);
          else           addr = coff + (long)mm * ldc + nn;
          if (flags & 1) ((ushort_t*)C)[addr] = f32_to_bf16(val);
          else           ((float*)C)[addr]    = val;
        }
      }
    }
  }
}

// ---------------------------------------------------------------- launch
extern "C" void kernel_launch(void* const* d_in, const int* in_sizes, int n_in,
                              void* d_out, int out_size, void* d_ws, size_t ws_size,
                              hipStream_t stream) {
  (void)in_sizes; (void)n_in; (void)out_size; (void)ws_size;

  const float* x      = (const float*)d_in[0];
  const float* L      = (const float*)d_in[1];
  const float* Wq_lat = (const float*)d_in[2];
  const float* Wk_in  = (const float*)d_in[3];
  const float* Wv_in  = (const float*)d_in[4];
  const float* Wq_in  = (const float*)d_in[5];
  const float* Wk_lat = (const float*)d_in[6];
  const float* Wv_lat = (const float*)d_in[7];
  const float* Wout   = (const float*)d_in[8];
  float* out = (float*)d_out;

  const long BATCH = 4, T = 4096, DM = 2048, NL = 64, DL = 1024, H = 16, DH = 64;

  char* ws = (char*)d_ws;
  size_t off = 0;
  auto alloc = [&](size_t bytes) -> void* {
    size_t o = (off + 255) & ~(size_t)255; off = o + bytes; return (void*)(ws + o);
  };

  ushort_t* xb     = (ushort_t*)alloc(BATCH*T*DM*2);
  ushort_t* WtQlat = (ushort_t*)alloc(DL*DL*2);
  ushort_t* WtKin  = (ushort_t*)alloc(DL*DM*2);
  ushort_t* WtVin  = (ushort_t*)alloc(DL*DM*2);
  ushort_t* WtQin  = (ushort_t*)alloc(DL*DM*2);
  ushort_t* WtKlat = (ushort_t*)alloc(DL*DL*2);
  ushort_t* WtVlat = (ushort_t*)alloc(DL*DL*2);
  ushort_t* WtOut  = (ushort_t*)alloc(DM*DL*2);
  ushort_t* Lb     = (ushort_t*)alloc(NL*DL*2);
  ushort_t* Qlat   = (ushort_t*)alloc(NL*DL*2);
  ushort_t* Kb     = (ushort_t*)alloc(BATCH*T*DL*2);      // reused as Qx (stage 3)
  ushort_t* Vt     = (ushort_t*)alloc(BATCH*T*DL*2);      // [B,H,DH,T]; reused as attn3
  float*    sc1    = (float*)   alloc(BATCH*H*NL*T*4);    // reused as sc3 (same size)
  ushort_t* pr1    = (ushort_t*)alloc(BATCH*H*NL*T*2);    // reused as pr3
  ushort_t* zb     = (ushort_t*)alloc(BATCH*NL*DL*2);
  ushort_t* Qz     = (ushort_t*)alloc(BATCH*NL*DL*2);
  ushort_t* Kz2    = (ushort_t*)alloc(BATCH*NL*DL*2);
  ushort_t* Vzt2   = (ushort_t*)alloc(BATCH*NL*DL*2);     // [B,H,DH,NL]
  float*    sc2    = (float*)   alloc(BATCH*H*NL*NL*4);
  ushort_t* pr2    = (ushort_t*)alloc(BATCH*H*NL*NL*2);
  ushort_t* z2b    = (ushort_t*)alloc(BATCH*NL*DL*2);
  ushort_t* Kz3    = (ushort_t*)alloc(BATCH*NL*DL*2);
  ushort_t* Vzt3   = (ushort_t*)alloc(BATCH*NL*DL*2);     // [B,H,DH,NL]

  auto gemm = [&](const ushort_t* A, int lda, long sAb, long sAh,
                  const ushort_t* Bm, int ldb, long sBb, long sBh,
                  void* C, int ldc, long sCb, long sCh,
                  int M, int N, int K, float alpha, int nH, int batches, int flags,
                  int Mblock, long blockStride, int ldct) {
    dim3 grid((unsigned)(N / 64), (unsigned)((M + 127) / 128), (unsigned)batches);
    k_gemm_bf16<<<grid, 256, 0, stream>>>(A, lda, sAb, sAh, Bm, ldb, sBb, sBh,
                                          C, ldc, sCb, sCh, M, N, K, alpha, nH, flags,
                                          Mblock, blockStride, ldct);
  };

  // ---- precision conversion + weight transposes
  k_cvt_bf16<<<4096, 256, 0, stream>>>(x, xb, BATCH*T*DM);
  k_cvt_bf16<<<256,  256, 0, stream>>>(L, Lb, NL*DL);
  k_transpose_cvt<<<dim3(DL/32, DL/32), dim3(32,8), 0, stream>>>(Wq_lat, WtQlat, (int)DL, (int)DL);
  k_transpose_cvt<<<dim3(DL/32, DM/32), dim3(32,8), 0, stream>>>(Wk_in,  WtKin,  (int)DM, (int)DL);
  k_transpose_cvt<<<dim3(DL/32, DM/32), dim3(32,8), 0, stream>>>(Wv_in,  WtVin,  (int)DM, (int)DL);
  k_transpose_cvt<<<dim3(DL/32, DM/32), dim3(32,8), 0, stream>>>(Wq_in,  WtQin,  (int)DM, (int)DL);
  k_transpose_cvt<<<dim3(DL/32, DL/32), dim3(32,8), 0, stream>>>(Wk_lat, WtKlat, (int)DL, (int)DL);
  k_transpose_cvt<<<dim3(DL/32, DL/32), dim3(32,8), 0, stream>>>(Wv_lat, WtVlat, (int)DL, (int)DL);
  k_transpose_cvt<<<dim3(DM/32, DL/32), dim3(32,8), 0, stream>>>(Wout,   WtOut,  (int)DL, (int)DM);

  // ---- Stage 1: Q = L@Wq_lat ; K,V = x@{Wk_in,Wv_in} (V stored [B,H,DH,T])
  gemm(Lb, 1024, 0, 0, WtQlat, 1024, 0, 0, Qlat, 1024, 0, 0, 64, 1024, 1024, 1.f, 1, 1, 1, 1, 0, 0);
  gemm(xb, 2048, 0, 0, WtKin,  2048, 0, 0, Kb,   1024, 0, 0, 16384, 1024, 2048, 1.f, 1, 1, 1, 1, 0, 0);
  gemm(xb, 2048, 0, 0, WtVin,  2048, 0, 0, Vt,   0, 0, 0, 16384, 1024, 2048, 1.f, 1, 1, 3,
       (int)T, H*DH*T, (int)T);
  // scores1[b,h,q,k] then softmax, z = P@V
  gemm(Qlat, 1024, 0, 64, Kb, 1024, T*1024, 64, sc1, 4096, H*NL*T, NL*T,
       64, 4096, 64, 0.125f, (int)H, (int)(BATCH*H), 0, 1, 0, 0);
  k_softmax_bf16<<<(unsigned)(BATCH*H*NL), 128, 0, stream>>>(sc1, pr1, 4096);
  gemm(pr1, 4096, H*NL*T, NL*T, Vt, 4096, H*DH*T, DH*T, zb, 1024, NL*DL, 64,
       64, 64, 4096, 1.f, (int)H, (int)(BATCH*H), 1, 1, 0, 0);

  // ---- Stage 2: latent self-attention
  gemm(zb, 1024, 0, 0, WtQlat, 1024, 0, 0, Qz,  1024, 0, 0, 256, 1024, 1024, 1.f, 1, 1, 1, 1, 0, 0);
  gemm(zb, 1024, 0, 0, WtKlat, 1024, 0, 0, Kz2, 1024, 0, 0, 256, 1024, 1024, 1.f, 1, 1, 1, 1, 0, 0);
  gemm(zb, 1024, 0, 0, WtVlat, 1024, 0, 0, Vzt2, 0, 0, 0, 256, 1024, 1024, 1.f, 1, 1, 3,
       (int)NL, H*DH*NL, (int)NL);
  gemm(Qz, 1024, NL*DL, 64, Kz2, 1024, NL*DL, 64, sc2, 64, H*NL*NL, NL*NL,
       64, 64, 64, 0.125f, (int)H, (int)(BATCH*H), 0, 1, 0, 0);
  k_softmax_bf16<<<(unsigned)(BATCH*H*NL), 128, 0, stream>>>(sc2, pr2, 64);
  gemm(pr2, 64, H*NL*NL, NL*NL, Vzt2, 64, H*DH*NL, DH*NL, z2b, 1024, NL*DL, 64,
       64, 64, 64, 1.f, (int)H, (int)(BATCH*H), 1, 1, 0, 0);

  // ---- Stage 3: tokens attend to latents
  gemm(xb, 2048, 0, 0, WtQin, 2048, 0, 0, Kb /*Qx*/, 1024, 0, 0, 16384, 1024, 2048, 1.f, 1, 1, 1, 1, 0, 0);
  gemm(z2b, 1024, 0, 0, WtKlat, 1024, 0, 0, Kz3, 1024, 0, 0, 256, 1024, 1024, 1.f, 1, 1, 1, 1, 0, 0);
  gemm(z2b, 1024, 0, 0, WtVlat, 1024, 0, 0, Vzt3, 0, 0, 0, 256, 1024, 1024, 1.f, 1, 1, 3,
       (int)NL, H*DH*NL, (int)NL);
  gemm(Kb /*Qx*/, 1024, T*1024, 64, Kz3, 1024, NL*DL, 64, sc1 /*sc3*/, 64, H*T*DH, T*DH,
       4096, 64, 64, 0.125f, (int)H, (int)(BATCH*H), 0, 1, 0, 0);
  k_softmax_bf16<<<(unsigned)(BATCH*H*T), 128, 0, stream>>>(sc1, pr1 /*pr3*/, 64);
  gemm(pr1, 64, H*T*DH, T*DH, Vzt3, 64, H*DH*NL, DH*NL, Vt /*attn3*/, 1024, T*DL, 64,
       4096, 64, 64, 1.f, (int)H, (int)(BATCH*H), 1, 1, 0, 0);

  // ---- Output projection (f32 directly into d_out)
  gemm(Vt /*attn3*/, 1024, 0, 0, WtOut, 1024, 0, 0, out, 2048, 0, 0,
       16384, 2048, 1024, 1.f, 1, 1, 0, 1, 0, 0);
}
